// ContTimeLSTM_7791070674955
// MI455X (gfx1250) — matrix operand order repaired and checked by reading, not compiled, not run on hardware
//
#include <hip/hip_runtime.h>
#include <stdint.h>

typedef __bf16 v16bf __attribute__((ext_vector_type(16)));
typedef float  v8f   __attribute__((ext_vector_type(8)));
typedef int    v4i   __attribute__((ext_vector_type(4)));
typedef __attribute__((address_space(1))) v4i* gv4i_p;   // global int4*
typedef __attribute__((address_space(3))) v4i* lv4i_p;   // LDS int4*

#define NL     512          // sequence length
#define NBATCH 32           // batch
#define NDIN   512          // input dim
#define NH     1024         // hidden
#define NK     (NDIN + NH)  // 1536
#define NBLK   64           // persistent workgroups
#define CH     (NH / NBLK)  // 16 hidden columns per block
#define NT     384          // 12 waves: (gate 0..5) x (M-tile 0..1)
#define S1     (3 * NH + 1) // per-step state row

#if __has_builtin(__builtin_amdgcn_global_load_async_to_lds_b128) && \
    __has_builtin(__builtin_amdgcn_s_wait_asynccnt)
#define ASYNC_STAGE 1
#else
#define ASYNC_STAGE 0
#endif

__device__ __forceinline__ unsigned short f2bf(float f) {
    unsigned u = __float_as_uint(f);
    unsigned r = (u + 0x7FFFu + ((u >> 16) & 1u)) >> 16;   // RNE
    return (unsigned short)r;
}
__device__ __forceinline__ float bf2f(unsigned short h) {
    return __uint_as_float(((unsigned)h) << 16);
}
// fast sigmoid: v_exp_f32 + v_rcp_f32 (no IEEE divide chain)
__device__ __forceinline__ float sigm(float x) {
    return __builtin_amdgcn_rcpf(1.0f + __expf(-x));
}
__device__ __forceinline__ float fast_tanh(float x) {
#if __has_builtin(__builtin_amdgcn_tanhf)
    return __builtin_amdgcn_tanhf(x);          // v_tanh_f32 (gfx1250 trans op)
#else
    return 2.0f * sigm(2.0f * x) - 1.0f;       // reference's own formula
#endif
}

// ---------- prep kernels ----------
__global__ void cvt_bf16_kernel(const float* __restrict__ src,
                                unsigned short* __restrict__ dst, long n) {
    long i = (long)blockIdx.x * blockDim.x + threadIdx.x;
    long stride = (long)gridDim.x * blockDim.x;
    for (; i < n; i += stride) dst[i] = f2bf(src[i]);
}
__global__ void zero_kernel(unsigned* __restrict__ p, int n) {
    int i = blockIdx.x * blockDim.x + threadIdx.x;
    if (i < n) p[i] = 0u;
}

// ---------- persistent scan kernel ----------
__global__ __launch_bounds__(NT) void ctlstm_kernel(
    const float* __restrict__ td,                 // (L,B,1)
    const float* __restrict__ o0,                 // (B,H)
    const float* __restrict__ cs0,                // (B,H)
    const float* __restrict__ ce0,                // (B,H)
    const float* __restrict__ d0,                 // (B,1)
    const float* __restrict__ bias,               // (6H+1)
    const float* __restrict__ dW,                 // (1536)
    const float* __restrict__ dB,                 // (1)
    const float* __restrict__ dBeta,              // (1)
    const unsigned short* __restrict__ xbf,       // (L,B,DIN) bf16
    const unsigned short* __restrict__ wbf,       // (6H,K) bf16
    unsigned short* __restrict__ hbuf,            // 2*B*H bf16 (double buffer)
    unsigned* __restrict__ cnt,                   // L barrier counters
    float* __restrict__ out)                      // outputs(B,L,H) ++ states(B,L,3H+1)
{
    __shared__ unsigned short sA[NBATCH * NK];          // x_s staged bf16 (96 KB)
    __shared__ float sO[NBATCH * CH], sCS[NBATCH * CH];
    __shared__ float sCE[NBATCH * CH], sC[NBATCH * CH];
    __shared__ float sProj[6 * NBATCH * CH];            // sigmoid(gates) (12 KB)
    __shared__ float sD[NBATCH];
    __shared__ float sPart[NT];

    const int tid = threadIdx.x;
    const int bid = blockIdx.x;
    const int j0  = bid * CH;
    const float beta  = dBeta[0];
    const float dbias = dB[0];

    float* outH = out;                                   // (B,L,H)
    float* outS = out + (size_t)NBATCH * NL * NH;        // (B,L,3H+1)

    // initial state load (block-local slice)
    for (int idx = tid; idx < NBATCH * CH; idx += NT) {
        int b = idx / CH, j = idx % CH, col = j0 + j;
        sO[idx]  = o0[b * NH + col];
        sCS[idx] = cs0[b * NH + col];
        sCE[idx] = ce0[b * NH + col];
    }
    if (tid < NBATCH) sD[tid] = d0[tid];
    __syncthreads();

    const int lane = tid & 31;
    const int wave = tid >> 5;
    const int g    = wave >> 1;      // gate 0..5
    const int m    = wave & 1;       // M-tile (batch rows 0-15 / 16-31)
    const int ln   = lane & 15;      // N within tile
    const int grp  = lane >> 4;
    // B fragment: column n = weight row (g*H + j0 + ln); lanes 0-15 K chunks [0..7],[16..23]; lanes 16-31 [8..15],[24..31]
    const unsigned short* wrow = wbf + ((size_t)g * NH + j0 + ln) * NK + grp * 8;
    // A fragment: row = batch m*16+ln; lanes 0-15 K chunks [0..7],[16..23]; lanes 16-31 [8..15],[24..31]
    const unsigned short* arow = sA + (m * 16 + ln) * NK;
    const float bv = bias[g * NH + j0 + ln];

    for (int t = 0; t < NL; ++t) {
        unsigned short* hcur = hbuf + (t & 1) * (NBATCH * NH);

        // ---- Phase A: c = cs + (ce-cs)*exp(-d*dt), h = o*tanh(c) ----
        for (int idx = tid; idx < NBATCH * CH; idx += NT) {
            int b = idx / CH, j = idx % CH, col = j0 + j;
            float dt = td[t * NBATCH + b];
            float c  = sCS[idx] + (sCE[idx] - sCS[idx]) * __expf(-sD[b] * dt);
            float h  = sO[idx] * fast_tanh(c);
            sC[idx] = c;
            hcur[b * NH + col] = f2bf(h);
            outH[(size_t)b * NL * NH + (size_t)t * NH + col] = h;
        }
        __syncthreads();

        // ---- grid barrier: all h(t) chunks visible ----
        if (tid == 0) {
            __hip_atomic_fetch_add(&cnt[t], 1u, __ATOMIC_ACQ_REL, __HIP_MEMORY_SCOPE_AGENT);
            while (__hip_atomic_load(&cnt[t], __ATOMIC_ACQUIRE, __HIP_MEMORY_SCOPE_AGENT) < NBLK)
                __builtin_amdgcn_s_sleep(1);
        }
        __syncthreads();

        // ---- stage x_s = [x_t | h(t)] into LDS (bf16) ----
#if ASYNC_STAGE
        {
            const unsigned short* xs = xbf + (size_t)t * NBATCH * NDIN;
            for (int u = tid; u < (NBATCH * NDIN) / 8; u += NT) {
                int b = u >> 6, r = u & 63;                 // DIN/8 = 64 per batch row
                __builtin_amdgcn_global_load_async_to_lds_b128(
                    (gv4i_p)(xs + u * 8),
                    (lv4i_p)(sA + b * NK + r * 8),
                    0, 0);
            }
            for (int u = tid; u < (NBATCH * NH) / 8; u += NT) {
                int b = u >> 7, r = u & 127;                // H/8 = 128 per batch row
                __builtin_amdgcn_global_load_async_to_lds_b128(
                    (gv4i_p)(hcur + u * 8),
                    (lv4i_p)(sA + b * NK + NDIN + r * 8),
                    0, 0);
            }
            __builtin_amdgcn_s_wait_asynccnt(0);
        }
#else
        {
            const uint4* xs = (const uint4*)(xbf + (size_t)t * NBATCH * NDIN);
            for (int u = tid; u < (NBATCH * NDIN) / 8; u += NT) {
                int b = u >> 6, r = u & 63;                 // DIN/8 = 64 per batch row
                ((uint4*)(sA + b * NK))[r] = xs[u];
            }
            const uint4* hs = (const uint4*)hcur;
            for (int u = tid; u < (NBATCH * NH) / 8; u += NT) {
                int b = u >> 7, r = u & 127;                // H/8 = 128 per batch row
                ((uint4*)(sA + b * NK + NDIN))[r] = hs[u];
            }
        }
#endif
        __syncthreads();

        // prefetch next step's x tile (32 KB) into cache while we do the GEMM
        if (t + 1 < NL) {
            const char* nx = (const char*)(xbf + (size_t)(t + 1) * NBATCH * NDIN);
            for (int u = tid; u < (NBATCH * NDIN * 2) / 256; u += NT)   // 128 lines
                __builtin_prefetch(nx + u * 256, 0, 1);
        }

        // ---- GEMM: 16 x 16 tile per wave, K=1536, bf16 WMMA ----
        v8f acc = {};
        #pragma unroll 4
        for (int kb = 0; kb < NK; kb += 32) {
            union { v16bf v; uint4 u[2]; } a, bm;
            a.u[0]  = *(const uint4*)(arow + kb + grp * 8);
            a.u[1]  = *(const uint4*)(arow + kb + 16 + grp * 8);
            bm.u[0] = *(const uint4*)(wrow + kb);
            bm.u[1] = *(const uint4*)(wrow + kb + 16);
            acc = __builtin_amdgcn_wmma_f32_16x16x32_bf16(
                false, a.v, false, bm.v, (short)0, acc, false, false);
        }
        #pragma unroll
        for (int r = 0; r < 8; ++r) {
            int brow = m * 16 + grp * 8 + r;                // batch row of this acc element
            sProj[g * (NBATCH * CH) + brow * CH + ln] = sigm(acc[r] + bv);
        }

        // ---- d_new partial dot: x_s . d_weight (each block recomputes) ----
        {
            int b = tid / 12, r = tid - b * 12;
            float part = 0.f;
            const unsigned short* xr = sA + b * NK + r * 128;
            const float* wr = dW + r * 128;
            #pragma unroll 8
            for (int k = 0; k < 128; ++k) part += bf2f(xr[k]) * wr[k];
            sPart[tid] = part;
        }
        __syncthreads();

        // ---- state update (block-local) + state outputs ----
        for (int idx = tid; idx < NBATCH * CH; idx += NT) {
            int b = idx / CH, j = idx % CH, col = j0 + j;
            float i_g = sProj[idx];
            float f_g = sProj[NBATCH * CH + idx];
            float ie  = sProj[2 * NBATCH * CH + idx];
            float fe  = sProj[3 * NBATCH * CH + idx];
            float z   = 2.f * sProj[4 * NBATCH * CH + idx] - 1.f;
            float on  = sProj[5 * NBATCH * CH + idx];
            float c   = sC[idx];
            float cs  = f_g * c + i_g * z;
            float ce  = fe * sCE[idx] + ie * z;
            sO[idx] = on; sCS[idx] = cs; sCE[idx] = ce;
            float* sb = outS + (size_t)b * NL * S1 + (size_t)t * S1;
            sb[col]          = on;
            sb[NH + col]     = cs;
            sb[2 * NH + col] = ce;
        }
        if (tid < NBATCH) {
            float s = dbias;
            #pragma unroll
            for (int q = 0; q < 12; ++q) s += sPart[tid * 12 + q];
            float by = beta * s;
            float dn = (by > 20.f) ? by : __logf(1.f + __expf(by));   // softplus, fast log
            dn /= beta;
            sD[tid] = dn;
            if (bid == 0)
                outS[(size_t)tid * NL * S1 + (size_t)t * S1 + 3 * NH] = dn;
        }
        __syncthreads();
    }
}

// ---------- launcher ----------
extern "C" void kernel_launch(void* const* d_in, const int* in_sizes, int n_in,
                              void* d_out, int out_size, void* d_ws, size_t ws_size,
                              hipStream_t stream) {
    const float* x      = (const float*)d_in[0];
    const float* td     = (const float*)d_in[1];
    const float* o0     = (const float*)d_in[2];
    const float* cs0    = (const float*)d_in[3];
    const float* ce0    = (const float*)d_in[4];
    const float* dst0   = (const float*)d_in[5];
    const float* weight = (const float*)d_in[6];
    const float* bias   = (const float*)d_in[7];
    const float* dW     = (const float*)d_in[8];
    const float* dB     = (const float*)d_in[9];
    const float* dBeta  = (const float*)d_in[10];
    float* out = (float*)d_out;

    char* ws = (char*)d_ws;
    const size_t xBytes = (size_t)NL * NBATCH * NDIN * 2;     // 16 MB bf16 x
    const size_t wBytes = (size_t)6 * NH * NK * 2;            // 18.9 MB bf16 weight
    const size_t hBytes = (size_t)2 * NBATCH * NH * 2;        // 128 KB h double buffer
    unsigned short* xbf  = (unsigned short*)ws;
    unsigned short* wbf  = (unsigned short*)(ws + xBytes);
    unsigned short* hbuf = (unsigned short*)(ws + xBytes + wBytes);
    unsigned* cnt        = (unsigned*)(ws + xBytes + wBytes + hBytes);

    cvt_bf16_kernel<<<2048, 256, 0, stream>>>(x, xbf, (long)NL * NBATCH * NDIN);
    cvt_bf16_kernel<<<2048, 256, 0, stream>>>(weight, wbf, (long)6 * NH * NK);
    zero_kernel<<<(NL + 255) / 256, 256, 0, stream>>>(cnt, NL);

    ctlstm_kernel<<<NBLK, NT, 0, stream>>>(td, o0, cs0, ce0, dst0, bias, dW, dB,
                                           dBeta, xbf, wbf, hbuf, cnt, out);
}
